// DCRNNEncoder_10273561772734
// MI455X (gfx1250) — matrix operand order repaired and checked by reading, not compiled, not run on hardware
//
#include <hip/hip_runtime.h>
#include <hip/hip_bf16.h>

// ---------------- problem dimensions ----------------
#define TT 32    // time steps
#define BB 32    // batch
#define NN 512   // nodes
#define DD 128   // input feature
#define HH 128   // hidden
#define LL 2     // layers
#define FF 256   // total = D + H
#define GG 768   // num_mat * total = 3 * 256

// ---------------- WMMA vector types ----------------
typedef __attribute__((ext_vector_type(16))) __bf16 v16bf;
typedef __attribute__((ext_vector_type(8)))  __bf16 v8bf;
typedef __attribute__((ext_vector_type(4)))  __bf16 v4bf;
typedef __attribute__((ext_vector_type(8)))  float  v8f;

// ---------------- tiling ----------------
#define TM 128    // block rows
#define TN 64     // block cols
#define TK 32     // k step
#define LDST 40   // padded LDS row stride (elements); 80B rows keep 16B alignment

// ============================================================
// small conversion / elementwise kernels
// ============================================================
__global__ void cvt_bf(const float* __restrict__ s, __bf16* __restrict__ d, int n) {
    int i = blockIdx.x * blockDim.x + threadIdx.x;
    if (i < n) d[i] = (__bf16)s[i];
}

// WT[o*K + g] = W[g*C + o]   (W is [K rows, C cols] f32, WT is [C, K] bf16)
__global__ void transpose_bf(const float* __restrict__ W, __bf16* __restrict__ WT,
                             int K, int C) {
    int idx = blockIdx.x * blockDim.x + threadIdx.x;
    if (idx >= K * C) return;
    int g = idx / C, o = idx % C;
    WT[(size_t)o * K + g] = (__bf16)W[(size_t)g * C + o];
}

// build cat = [x_t, h] (useR=0) or [x_t, r*h] (useR=1) into hcat slice 0, 4 elems/thread
__global__ void build_cat(const float* __restrict__ xt, const float* __restrict__ h,
                          const float* __restrict__ gates, __bf16* __restrict__ hcat,
                          int useR) {
    int idx = blockIdx.x * blockDim.x + threadIdx.x;   // over B*N*F/4
    if (idx >= BB * NN * FF / 4) return;
    int row = idx >> 6;          // (b,n) flattened
    int o4  = (idx & 63) * 4;    // never straddles the 128 boundary
    float4 t;
    if (o4 < HH) {
        t = *(const float4*)(xt + (size_t)row * DD + o4);
    } else {
        t = *(const float4*)(h + (size_t)row * HH + (o4 - HH));
        if (useR) {
            float4 g = *(const float4*)(gates + (size_t)row * (2 * HH) + (o4 - HH));
            t.x *= g.x; t.y *= g.y; t.z *= g.z; t.w *= g.w;
        }
    }
    v4bf p;
    p[0] = (__bf16)t.x; p[1] = (__bf16)t.y; p[2] = (__bf16)t.z; p[3] = (__bf16)t.w;
    *(v4bf*)(hcat + (size_t)row * GG + o4) = p;
}

// ============================================================
// diffusion GEMM: per batch  Y = S * X   (mode 1)
//                         or Y = 2*S*X - X0 (mode 2, Chebyshev)
// Block tile 128x64, 8 waves, wave tile 16x64.
// LDS double-buffered; global loads for step k+1 staged in registers
// while WMMAs for step k run; one barrier per k-step.
// ============================================================
__global__ __launch_bounds__(256)
void gemm_diff(const __bf16* __restrict__ S, __bf16* __restrict__ hcat,
               int inOff, int outOff, int mode) {
    __shared__ __align__(16) __bf16 ldsA[2][TM * LDST];
    __shared__ __align__(16) __bf16 ldsB[2][TN * LDST];

    const int tilesPerBatch = (NN / TM) * (FF / TN);   // 16
    int b   = blockIdx.x / tilesPerBatch;
    int rem = blockIdx.x % tilesPerBatch;
    int m0 = (rem / (FF / TN)) * TM;
    int f0 = (rem % (FF / TN)) * TN;

    const __bf16* X  = hcat + (size_t)b * NN * GG + inOff;
    __bf16*       Y  = hcat + (size_t)b * NN * GG + outOff;
    const __bf16* X0 = hcat + (size_t)b * NN * GG;     // slice 0 (for mode 2)

    int tid  = threadIdx.x;
    int lane = tid & 31;
    int wv   = tid >> 5;                               // 8 row tiles of 16

    int ar = tid >> 1, ac = (tid & 1) * 16;            // A: 16 elems per thread
    int bf = (tid & 15) * 4, bk = (tid >> 4) * 2;      // B: 2 k-rows x 4 f per thread

    // fragment LDS coordinates
    int am  = wv * 16 + (lane & 15);
    int akb = (lane >> 4) * 16;
    int bn  = lane & 15;
    int bkb = (lane >> 4) * 16;

    v8f acc0 = {}, acc1 = {}, acc2 = {}, acc3 = {};
    v8bf a0, a1; v4bf b0, b1;

    auto load_global = [&](int k0) {
        const __bf16* As = S + (size_t)(m0 + ar) * NN + k0 + ac;
        a0 = *(const v8bf*)(As);
        a1 = *(const v8bf*)(As + 8);
        const __bf16* Bs = X + (size_t)(k0 + bk) * GG + f0 + bf;
        b0 = *(const v4bf*)(Bs);
        b1 = *(const v4bf*)(Bs + GG);
    };
    auto store_lds = [&](int buf) {
        *(v8bf*)&ldsA[buf][ar * LDST + ac]     = a0;
        *(v8bf*)&ldsA[buf][ar * LDST + ac + 8] = a1;
#pragma unroll
        for (int j = 0; j < 4; ++j) {
            union { __bf16 e[2]; unsigned u; } p;
            p.e[0] = b0[j]; p.e[1] = b1[j];
            *(unsigned*)&ldsB[buf][(bf + j) * LDST + bk] = p.u;
        }
    };

    load_global(0);
    store_lds(0);
    __syncthreads();

    const int nsteps = NN / TK;
    for (int s = 0; s < nsteps; ++s) {
        int cur = s & 1;
        if (s + 1 < nsteps) load_global((s + 1) * TK);
        if (s + 2 < nsteps) {
            __builtin_prefetch(S + (size_t)(m0 + ar) * NN + (s + 2) * TK + ac, 0, 3);
            __builtin_prefetch(X + (size_t)((s + 2) * TK + bk) * GG + f0 + bf, 0, 3);
        }

        union { v16bf v; v8bf h[2]; } af, bfr0, bfr1, bfr2, bfr3;
        af.h[0] = *(const v8bf*)&ldsA[cur][am * LDST + akb];
        af.h[1] = *(const v8bf*)&ldsA[cur][am * LDST + akb + 8];
        bfr0.h[0] = *(const v8bf*)&ldsB[cur][(bn)      * LDST + bkb];
        bfr0.h[1] = *(const v8bf*)&ldsB[cur][(bn)      * LDST + bkb + 8];
        bfr1.h[0] = *(const v8bf*)&ldsB[cur][(bn + 16) * LDST + bkb];
        bfr1.h[1] = *(const v8bf*)&ldsB[cur][(bn + 16) * LDST + bkb + 8];
        bfr2.h[0] = *(const v8bf*)&ldsB[cur][(bn + 32) * LDST + bkb];
        bfr2.h[1] = *(const v8bf*)&ldsB[cur][(bn + 32) * LDST + bkb + 8];
        bfr3.h[0] = *(const v8bf*)&ldsB[cur][(bn + 48) * LDST + bkb];
        bfr3.h[1] = *(const v8bf*)&ldsB[cur][(bn + 48) * LDST + bkb + 8];

        acc0 = __builtin_amdgcn_wmma_f32_16x16x32_bf16(false, af.v, false, bfr0.v, (short)0, acc0, false, false);
        acc1 = __builtin_amdgcn_wmma_f32_16x16x32_bf16(false, af.v, false, bfr1.v, (short)0, acc1, false, false);
        acc2 = __builtin_amdgcn_wmma_f32_16x16x32_bf16(false, af.v, false, bfr2.v, (short)0, acc2, false, false);
        acc3 = __builtin_amdgcn_wmma_f32_16x16x32_bf16(false, af.v, false, bfr3.v, (short)0, acc3, false, false);

        if (s + 1 < nsteps) store_lds(1 - cur);
        __syncthreads();
    }

    // --- epilogue ---
    int mBase = m0 + wv * 16 + ((lane >> 4) << 3);
    v8f* accs[4] = { &acc0, &acc1, &acc2, &acc3 };
#pragma unroll
    for (int ss = 0; ss < 4; ++ss) {
        int ng = f0 + ss * 16 + bn;
#pragma unroll
        for (int r = 0; r < 8; ++r) {
            int mg = mBase + r;
            float v = (*accs[ss])[r];
            if (mode == 2) v = 2.0f * v - (float)X0[(size_t)mg * GG + ng];
            Y[(size_t)mg * GG + ng] = (__bf16)v;
        }
    }
}

// ============================================================
// projection GEMM: out = act( hcat[16384, 768] x W[768, O] + bias )
// WT is pre-transposed bf16 [O, 768] (no LDS transpose needed).
// mode 0 (O=256): sigmoid -> gates buffer
// mode 1 (O=128): c = tanh(.); h = u*h + (1-u)*c; write h and outs
// ============================================================
__global__ __launch_bounds__(256)
void gemm_proj(const __bf16* __restrict__ Acat, const __bf16* __restrict__ WT,
               const float* __restrict__ bias, float* __restrict__ gates,
               float* __restrict__ hstate, float* __restrict__ outs,
               int Ocols, int mode) {
    __shared__ __align__(16) __bf16 ldsA[2][TM * LDST];
    __shared__ __align__(16) __bf16 ldsB[2][TN * LDST];

    int cb = Ocols / TN;                   // 4 or 2 column blocks
    int m0 = (blockIdx.x / cb) * TM;       // row block over B*N = 16384
    int o0 = (blockIdx.x % cb) * TN;

    int tid  = threadIdx.x;
    int lane = tid & 31;
    int wv   = tid >> 5;

    int ar = tid >> 1, ac = (tid & 1) * 16;    // A: 16 elems per thread (128 rows)
    int br = tid >> 2, bc = (tid & 3) * 8;     // B: 8 elems per thread (64 rows)

    int am  = wv * 16 + (lane & 15);
    int akb = (lane >> 4) * 16;
    int bn  = lane & 15;
    int bkb = (lane >> 4) * 16;

    v8f acc0 = {}, acc1 = {}, acc2 = {}, acc3 = {};
    v8bf a0, a1, bq;

    auto load_global = [&](int k0) {
        const __bf16* As = Acat + (size_t)(m0 + ar) * GG + k0 + ac;
        a0 = *(const v8bf*)(As);
        a1 = *(const v8bf*)(As + 8);
        bq = *(const v8bf*)(WT + (size_t)(o0 + br) * GG + k0 + bc);
    };
    auto store_lds = [&](int buf) {
        *(v8bf*)&ldsA[buf][ar * LDST + ac]     = a0;
        *(v8bf*)&ldsA[buf][ar * LDST + ac + 8] = a1;
        *(v8bf*)&ldsB[buf][br * LDST + bc]     = bq;
    };

    load_global(0);
    store_lds(0);
    __syncthreads();

    const int nsteps = GG / TK;
    for (int s = 0; s < nsteps; ++s) {
        int cur = s & 1;
        if (s + 1 < nsteps) load_global((s + 1) * TK);
        if (s + 2 < nsteps) {
            __builtin_prefetch(Acat + (size_t)(m0 + ar) * GG + (s + 2) * TK + ac, 0, 3);
            __builtin_prefetch(WT + (size_t)(o0 + br) * GG + (s + 2) * TK + bc, 0, 3);
        }

        union { v16bf v; v8bf h[2]; } af, bfr0, bfr1, bfr2, bfr3;
        af.h[0] = *(const v8bf*)&ldsA[cur][am * LDST + akb];
        af.h[1] = *(const v8bf*)&ldsA[cur][am * LDST + akb + 8];
        bfr0.h[0] = *(const v8bf*)&ldsB[cur][(bn)      * LDST + bkb];
        bfr0.h[1] = *(const v8bf*)&ldsB[cur][(bn)      * LDST + bkb + 8];
        bfr1.h[0] = *(const v8bf*)&ldsB[cur][(bn + 16) * LDST + bkb];
        bfr1.h[1] = *(const v8bf*)&ldsB[cur][(bn + 16) * LDST + bkb + 8];
        bfr2.h[0] = *(const v8bf*)&ldsB[cur][(bn + 32) * LDST + bkb];
        bfr2.h[1] = *(const v8bf*)&ldsB[cur][(bn + 32) * LDST + bkb + 8];
        bfr3.h[0] = *(const v8bf*)&ldsB[cur][(bn + 48) * LDST + bkb];
        bfr3.h[1] = *(const v8bf*)&ldsB[cur][(bn + 48) * LDST + bkb + 8];

        acc0 = __builtin_amdgcn_wmma_f32_16x16x32_bf16(false, af.v, false, bfr0.v, (short)0, acc0, false, false);
        acc1 = __builtin_amdgcn_wmma_f32_16x16x32_bf16(false, af.v, false, bfr1.v, (short)0, acc1, false, false);
        acc2 = __builtin_amdgcn_wmma_f32_16x16x32_bf16(false, af.v, false, bfr2.v, (short)0, acc2, false, false);
        acc3 = __builtin_amdgcn_wmma_f32_16x16x32_bf16(false, af.v, false, bfr3.v, (short)0, acc3, false, false);

        if (s + 1 < nsteps) store_lds(1 - cur);
        __syncthreads();
    }

    int mBase = m0 + wv * 16 + ((lane >> 4) << 3);
    v8f* accs[4] = { &acc0, &acc1, &acc2, &acc3 };
#pragma unroll
    for (int ss = 0; ss < 4; ++ss) {
        int o = o0 + ss * 16 + bn;
        float bsv = bias[o];
#pragma unroll
        for (int r = 0; r < 8; ++r) {
            int row = mBase + r;
            float v = (*accs[ss])[r] + bsv;
            if (mode == 0) {
                gates[(size_t)row * (2 * HH) + o] = 1.0f / (1.0f + expf(-v));
            } else {
                float c = tanhf(v);
                float u = gates[(size_t)row * (2 * HH) + HH + o];
                float h = hstate[(size_t)row * HH + o];
                float hn = u * h + (1.0f - u) * c;
                hstate[(size_t)row * HH + o] = hn;
                outs[(size_t)row * HH + o] = hn;
            }
        }
    }
}

// ============================================================
// host-side orchestration
// ============================================================
extern "C" void kernel_launch(void* const* d_in, const int* in_sizes, int n_in,
                              void* d_out, int out_size, void* d_ws, size_t ws_size,
                              hipStream_t stream) {
    (void)in_sizes; (void)n_in; (void)out_size; (void)ws_size;

    const float* inputs = (const float*)d_in[0];   // [T,B,N,D]
    const float* h0     = (const float*)d_in[1];   // [L,B,N,H]
    const float* sup    = (const float*)d_in[2];   // [1,N,N]
    const float* Wg     = (const float*)d_in[3];   // [L,768,256]
    const float* bg     = (const float*)d_in[4];   // [L,256]
    const float* Wc     = (const float*)d_in[5];   // [L,768,128]
    const float* bc     = (const float*)d_in[6];   // [L,128]

    float* out    = (float*)d_out;
    float* finals = out;                             // [L,B,N,H]
    float* cur    = out + (size_t)LL * BB * NN * HH; // [T,B,N,H] (layer-0 -> layer-1 buffer)

    // workspace carve-up
    char* w = (char*)d_ws;
    __bf16* Sbf = (__bf16*)w; w += (size_t)NN * NN * 2;
    __bf16* WgT = (__bf16*)w; w += (size_t)LL * FF * GG * 2;
    __bf16* WcT = (__bf16*)w; w += (size_t)LL * HH * GG * 2;
    float*  hst = (float*)w;  w += (size_t)BB * NN * HH * 4;
    float*  gat = (float*)w;  w += (size_t)BB * NN * (2 * HH) * 4;
    __bf16* hcat = (__bf16*)w; // [B, N, 768] bf16

    const size_t BNH = (size_t)BB * NN * HH;

    // one-time conversions
    cvt_bf<<<(NN * NN + 255) / 256, 256, 0, stream>>>(sup, Sbf, NN * NN);
    for (int l = 0; l < LL; ++l) {
        transpose_bf<<<(GG * FF + 255) / 256, 256, 0, stream>>>(
            Wg + (size_t)l * GG * FF, WgT + (size_t)l * FF * GG, GG, FF);
        transpose_bf<<<(GG * HH + 255) / 256, 256, 0, stream>>>(
            Wc + (size_t)l * GG * HH, WcT + (size_t)l * HH * GG, GG, HH);
    }

    const int diffBlocks  = BB * (NN / TM) * (FF / TN);            // 512
    const int projBlocksG = (BB * NN / TM) * ((2 * HH) / TN);      // 512
    const int projBlocksC = (BB * NN / TM) * (HH / TN);            // 256
    const int catBlocks   = (BB * NN * FF / 4) / 256;              // 4096

    for (int l = 0; l < LL; ++l) {
        hipMemcpyAsync(hst, h0 + (size_t)l * BNH, BNH * sizeof(float),
                       hipMemcpyDeviceToDevice, stream);
        for (int t = 0; t < TT; ++t) {
            const float* xt = (l == 0) ? inputs + (size_t)t * BB * NN * DD
                                       : cur + (size_t)t * BNH;
            // gates path
            build_cat<<<catBlocks, 256, 0, stream>>>(xt, hst, gat, hcat, 0);
            gemm_diff<<<diffBlocks, 256, 0, stream>>>(Sbf, hcat, 0, FF, 1);
            gemm_diff<<<diffBlocks, 256, 0, stream>>>(Sbf, hcat, FF, 2 * FF, 2);
            gemm_proj<<<projBlocksG, 256, 0, stream>>>(
                hcat, WgT + (size_t)l * FF * GG, bg + (size_t)l * 2 * HH,
                gat, nullptr, nullptr, 2 * HH, 0);
            // candidate path + GRU update
            build_cat<<<catBlocks, 256, 0, stream>>>(xt, hst, gat, hcat, 1);
            gemm_diff<<<diffBlocks, 256, 0, stream>>>(Sbf, hcat, 0, FF, 1);
            gemm_diff<<<diffBlocks, 256, 0, stream>>>(Sbf, hcat, FF, 2 * FF, 2);
            gemm_proj<<<projBlocksC, 256, 0, stream>>>(
                hcat, WcT + (size_t)l * HH * GG, bc + (size_t)l * HH,
                gat, hst, cur + (size_t)t * BNH, HH, 1);
        }
        hipMemcpyAsync(finals + (size_t)l * BNH, hst, BNH * sizeof(float),
                       hipMemcpyDeviceToDevice, stream);
    }
}
